// GraphAttentionLayer_35218731827948
// MI455X (gfx1250) — matrix operand order repaired and checked by reading, not compiled, not run on hardware
//
#include <hip/hip_runtime.h>
#include <hip/hip_bf16.h>

#define N_ROWS 8192
#define F_IN   512
#define F_OUT  256
#define NEG_INF -1.0e9f

typedef __attribute__((ext_vector_type(16))) __bf16 v16bf;
typedef __attribute__((ext_vector_type(8)))  float  v8f;

union BF16x16 { v16bf v; unsigned u[8]; };

// Hardware packed f32x2 -> bf16x2 convert (RNE). One VOP3 instead of the
// ~10-op software RNE sequence per pair.
__device__ __forceinline__ unsigned pack_bf16(float lo, float hi) {
    unsigned d;
    asm("v_cvt_pk_bf16_f32 %0, %1, %2" : "=v"(d) : "v"(lo), "v"(hi));
    return d;
}
__device__ __forceinline__ float bf16_to_f32(unsigned short h) {
    return __uint_as_float(((unsigned)h) << 16);
}

// CDNA5 async global->LDS copy (ASYNCcnt-tracked), per ISA 15.18.3 op 98.
// vdst = per-lane LDS byte address, vaddr = per-lane 64-bit global address.
__device__ __forceinline__ void async_load_b128(void* lds_dst, const void* gsrc) {
    unsigned lds_off = (unsigned)(size_t)lds_dst;   // addr[31:0] == LDS offset
    asm volatile("global_load_async_to_lds_b128 %0, %1, off"
                 :: "v"(lds_off), "v"(gsrc) : "memory");
}
__device__ __forceinline__ void wait_async0() {
    asm volatile("s_wait_asynccnt 0x0" ::: "memory");
}

// ---------------------------------------------------------------------------
// Kernel 1: h = x @ W (8192x512 @ 512x256), bf16 WMMA, f32 accumulate.
// h stored TRANSPOSED bf16: h_t[n * 8192 + row] so the attention kernel's
// B-operand {K,K+1} pairs (consecutive j) are contiguous b32 loads.
// Block = 128 threads (4 waves): rows [rbase,rbase+16) x all 256 cols,
// wave w owns cols [64w,64w+64) as 4 WMMA accumulators.
// LDS tiles are stored as packed bf16 pairs (one dword = {K,K+1}).
// ---------------------------------------------------------------------------
__global__ __launch_bounds__(128) void gat_hgemm(const float* __restrict__ x,
                                                 const float* __restrict__ W,
                                                 unsigned short* __restrict__ h_t) {
    __shared__ unsigned a_lds[16 * 16];     // x tile  [m][k/2] packed bf16x2
    __shared__ unsigned b_lds[256 * 16];    // W tile  [n][k/2] packed bf16x2

    const int rbase = blockIdx.x * 16;
    const int t    = threadIdx.x;
    const int lane = t & 31;
    const int wave = t >> 5;
    const int half = lane >> 4;
    const int m    = lane & 15;

    v8f acc[4];
    #pragma unroll
    for (int s = 0; s < 4; ++s)
        #pragma unroll
        for (int r = 0; r < 8; ++r) acc[s][r] = 0.0f;

    for (int kk = 0; kk < F_IN; kk += 32) {
        // Stage A: each thread loads 4 consecutive x floats (b128) -> 2 packed dwords
        {
            int r  = t >> 3;            // row 0..15
            int c0 = (t * 4) & 31;      // k offset 0,4,...,28
            const float4 xv = *(const float4*)&x[(rbase + r) * F_IN + kk + c0];
            a_lds[t * 2 + 0] = pack_bf16(xv.x, xv.y);
            a_lds[t * 2 + 1] = pack_bf16(xv.z, xv.w);
        }
        // Stage B: W[kk..kk+32][0..256] -> b_lds[n][k2]; coalesced over n
        #pragma unroll
        for (int i = 0; i < 32; ++i) {
            int j  = t + i * 128;
            int n  = j & 255;
            int k2 = j >> 8;            // 0..15
            float w0 = W[(kk + 2 * k2    ) * F_OUT + n];
            float w1 = W[(kk + 2 * k2 + 1) * F_OUT + n];
            b_lds[n * 16 + k2] = pack_bf16(w0, w1);
        }
        __syncthreads();

        // A operand (16-bit 16x32): VGPR v -> k2 = (v/4)*8 + half*4 + (v&3)
        BF16x16 A;
        #pragma unroll
        for (int v = 0; v < 8; ++v) {
            int k2 = ((v >> 2) << 3) + (half << 2) + (v & 3);
            A.u[v] = a_lds[m * 16 + k2];
        }
        // B operand (16-bit 32x16): VGPR v -> k2 = (v/4)*8 + half*4 + (v&3), same as A
        #pragma unroll
        for (int s = 0; s < 4; ++s) {
            int n = wave * 64 + s * 16 + m;
            BF16x16 B;
            #pragma unroll
            for (int v = 0; v < 8; ++v)
                B.u[v] = b_lds[n * 16 + ((v >> 2) << 3) + (half << 2) + (v & 3)];
            acc[s] = __builtin_amdgcn_wmma_f32_16x16x32_bf16(
                false, A.v, false, B.v, (short)0, acc[s], false, false);
        }
        __syncthreads();
    }

    // Packed bf16 writeback, transposed: rows r,r+1 are consecutive in h_t
    #pragma unroll
    for (int s = 0; s < 4; ++s) {
        int n = wave * 64 + s * 16 + m;
        unsigned* h32 = (unsigned*)&h_t[n * N_ROWS + rbase + half * 8];
        #pragma unroll
        for (int r = 0; r < 4; ++r)
            h32[r] = pack_bf16(acc[s][2 * r], acc[s][2 * r + 1]);
    }
}

// ---------------------------------------------------------------------------
// Kernel 2: e_i = h @ a_src, e_j = h @ a_dst; coalesced over transposed h_t.
// ---------------------------------------------------------------------------
__global__ __launch_bounds__(256) void gat_evec(const unsigned short* __restrict__ h_t,
                                                const float* __restrict__ a,
                                                float* __restrict__ e_i,
                                                float* __restrict__ e_j) {
    int row = blockIdx.x * 256 + threadIdx.x;
    float ei = 0.0f, ej = 0.0f;
    for (int n = 0; n < F_OUT; ++n) {
        float hv = bf16_to_f32(h_t[n * N_ROWS + row]);
        ei += hv * a[n];
        ej += hv * a[F_OUT + n];
    }
    e_i[row] = ei;
    e_j[row] = ej;
}

// ---------------------------------------------------------------------------
// Kernel 3: fused masked-softmax attention, flash-style online softmax.
// out = elu( softmax(mask(lrelu(e_i + e_j^T), adj)) @ h )
// adj (256 MB, the HBM roofline) is streamed EXACTLY ONCE via double-buffered
// GLOBAL_LOAD_ASYNC_TO_LDS_B128 prefetch: tile jb+32 is in flight while tile
// jb is scored / softmaxed / WMMA-accumulated. Each lane loads the 16B it
// later reads back, so s_wait_asynccnt 0 alone orders producer->consumer.
// ---------------------------------------------------------------------------
__global__ __launch_bounds__(128) void gat_attn(const int* __restrict__ adj,
                                                const unsigned short* __restrict__ h_t,
                                                const float* __restrict__ e_i,
                                                const float* __restrict__ e_j,
                                                float* __restrict__ out) {
    __shared__ int      adj_lds[2][16 * 32];   // double-buffered adj tile
    __shared__ float    sbuf[16 * 32];         // scores, then p for row sums
    __shared__ unsigned p_lds[16 * 16];        // P tile [m][k/2] packed bf16x2
    __shared__ float mrow[16];                 // running row max
    __shared__ float lrow[16];                 // running row sum
    __shared__ float srow[16];                 // per-tile rescale factor
    __shared__ float eirow[16];

    const int rbase = blockIdx.x * 16;
    const int t    = threadIdx.x;
    const int lane = t & 31;
    const int wave = t >> 5;
    const int half = lane >> 4;
    const int m    = lane & 15;
    const int r    = t >> 3;                   // this thread's score row
    const int c0   = (t * 4) & 31;             // this thread's 4-col offset

    if (t < 16) {
        mrow[t]  = -1.0e30f;
        lrow[t]  = 0.0f;
        eirow[t] = e_i[rbase + t];
    }
    v8f acc[4];
    #pragma unroll
    for (int s = 0; s < 4; ++s)
        #pragma unroll
        for (int q = 0; q < 8; ++q) acc[s][q] = 0.0f;

    // Prologue: async-prefetch tile 0 into buffer 0
    async_load_b128(&adj_lds[0][t * 4], &adj[(size_t)(rbase + r) * N_ROWS + c0]);
    __syncthreads();

    for (int jb = 0; jb < N_ROWS; jb += 32) {
        const int buf = (jb >> 5) & 1;

        wait_async0();                                   // tile jb resident
        if (jb + 32 < N_ROWS)                            // launch tile jb+32
            async_load_b128(&adj_lds[buf ^ 1][t * 4],
                            &adj[(size_t)(rbase + r) * N_ROWS + jb + 32 + c0]);

        // ---- scores: s = mask(lrelu(e_i + e_j)) ----
        {
            const float4 ejv = *(const float4*)&e_j[jb + c0];
            float ei0 = eirow[r];
            float s0 = ei0 + ejv.x, s1 = ei0 + ejv.y, s2 = ei0 + ejv.z, s3 = ei0 + ejv.w;
            s0 = s0 > 0.0f ? s0 : 0.2f * s0;
            s1 = s1 > 0.0f ? s1 : 0.2f * s1;
            s2 = s2 > 0.0f ? s2 : 0.2f * s2;
            s3 = s3 > 0.0f ? s3 : 0.2f * s3;
            const int* at = &adj_lds[buf][t * 4];
            float4 sv;
            sv.x = at[0] > 0 ? s0 : NEG_INF;
            sv.y = at[1] > 0 ? s1 : NEG_INF;
            sv.z = at[2] > 0 ? s2 : NEG_INF;
            sv.w = at[3] > 0 ? s3 : NEG_INF;
            *(float4*)&sbuf[t * 4] = sv;
        }
        __syncthreads();

        // ---- online-softmax: tile max, new max, rescale factor ----
        if (t < 16) {
            float mt = -1.0e30f;
            #pragma unroll
            for (int c = 0; c < 32; ++c) mt = fmaxf(mt, sbuf[t * 32 + c]);
            float mold = mrow[t];
            float mnew = fmaxf(mold, mt);
            float sc   = __expf(mold - mnew);
            srow[t] = sc;
            mrow[t] = mnew;
            lrow[t] = lrow[t] * sc;
        }
        __syncthreads();

        // ---- p = exp(s - mnew); packed bf16 into WMMA A layout ----
        {
            float4 sv = *(const float4*)&sbuf[t * 4];
            float mr = mrow[r];
            float p0 = __expf(sv.x - mr), p1 = __expf(sv.y - mr);
            float p2 = __expf(sv.z - mr), p3 = __expf(sv.w - mr);
            *(float4*)&sbuf[t * 4] = make_float4(p0, p1, p2, p3);
            p_lds[t * 2 + 0] = pack_bf16(p0, p1);
            p_lds[t * 2 + 1] = pack_bf16(p2, p3);
        }
        __syncthreads();

        if (t < 16) {
            float ssum = 0.0f;
            #pragma unroll
            for (int c = 0; c < 32; ++c) ssum += sbuf[t * 32 + c];
            lrow[t] += ssum;
        }

        // ---- rescale accumulators by exp(mold - mnew) ----
        float scv[8];
        #pragma unroll
        for (int q = 0; q < 8; ++q) scv[q] = srow[half * 8 + q];
        #pragma unroll
        for (int s = 0; s < 4; ++s)
            #pragma unroll
            for (int q = 0; q < 8; ++q) acc[s][q] *= scv[q];

        // ---- acc += P(16x32) @ h(32x16 per subtile), bf16 WMMA ----
        BF16x16 A;
        #pragma unroll
        for (int v = 0; v < 8; ++v) {
            int k2 = ((v >> 2) << 3) + (half << 2) + (v & 3);
            A.u[v] = p_lds[m * 16 + k2];
        }
        #pragma unroll
        for (int s = 0; s < 4; ++s) {
            int n = wave * 64 + s * 16 + m;
            BF16x16 B;
            #pragma unroll
            for (int v = 0; v < 8; ++v)     // consecutive j contiguous in h_t
                B.u[v] = *(const unsigned*)&h_t[n * N_ROWS + jb + ((v >> 2) << 4) + (half << 3) + 2 * (v & 3)];
            acc[s] = __builtin_amdgcn_wmma_f32_16x16x32_bf16(
                false, A.v, false, B.v, (short)0, acc[s], false, false);
        }
        __syncthreads();
    }

    // ---- finalize: out = elu(acc / l) ----
    float linv[8];
    #pragma unroll
    for (int q = 0; q < 8; ++q) linv[q] = 1.0f / lrow[half * 8 + q];
    #pragma unroll
    for (int s = 0; s < 4; ++s) {
        int n = wave * 64 + s * 16 + m;
        #pragma unroll
        for (int q = 0; q < 8; ++q) {
            float o = acc[s][q] * linv[q];
            o = o > 0.0f ? o : (__expf(o) - 1.0f);
            out[(rbase + half * 8 + q) * F_OUT + n] = o;
        }
    }
}

// ---------------------------------------------------------------------------
// Launch: inputs = {x (8192x512 f32), adj (8192x8192 i32), W (512x256 f32),
//                   a (512x1 f32)}; output = 8192x256 f32.
// Workspace: h_t bf16 (4 MB) + e_i/e_j f32 (64 KB).
// ---------------------------------------------------------------------------
extern "C" void kernel_launch(void* const* d_in, const int* in_sizes, int n_in,
                              void* d_out, int out_size, void* d_ws, size_t ws_size,
                              hipStream_t stream) {
    const float* x   = (const float*)d_in[0];
    const int*   adj = (const int*)d_in[1];
    const float* W   = (const float*)d_in[2];
    const float* a   = (const float*)d_in[3];
    float*       out = (float*)d_out;

    unsigned short* h_t = (unsigned short*)d_ws;
    float* e_i = (float*)((char*)d_ws + (size_t)F_OUT * N_ROWS * sizeof(unsigned short));
    float* e_j = e_i + N_ROWS;

    gat_hgemm<<<N_ROWS / 16, 128, 0, stream>>>(x, W, h_t);
    gat_evec <<<N_ROWS / 256, 256, 0, stream>>>(h_t, a, e_i, e_j);
    gat_attn <<<N_ROWS / 16, 128, 0, stream>>>(adj, h_t, e_i, e_j, out);
}
